// GaudiFP8Linear_36507222016386
// MI455X (gfx1250) — matrix operand order repaired and checked by reading, not compiled, not run on hardware
//
#include <hip/hip_runtime.h>

// Problem constants (from reference): input [1,512,7168] f32, weight [18432,7168] f32,
// weight_scale_inv [144,56] f32, output [1,512,18432] f32.
#define M_DIM 512
#define K_DIM 7168
#define N_DIM 18432
#define SN 56            // K_DIM / 128
#define BM 128
#define BN 128
#define BK 32
#define KTILES (K_DIM / BK)      // 224
#define LDS_ROW_U 20             // uints per LDS row: 32 halfs data + 8 halfs pad = 40 halfs

typedef __attribute__((ext_vector_type(16))) _Float16 v16h;
typedef __attribute__((ext_vector_type(8)))  float    v8f;

union FragU {
    uint4 q[2];
    v16h  v;
};

__device__ __forceinline__ unsigned int pkh(float a, float b) {
    auto h = __builtin_amdgcn_cvt_pkrtz(a, b);   // v_cvt_pk_f16_f32 (RTZ) -> packed 2x__fp16
    return __builtin_bit_cast(unsigned int, h);
}

__device__ __forceinline__ void load16(const float* __restrict__ p, float4 r[4]) {
    const float4* p4 = (const float4*)p;       // 16B-aligned: k offsets are multiples of 16 floats
    r[0] = p4[0]; r[1] = p4[1]; r[2] = p4[2]; r[3] = p4[3];
}

__device__ __forceinline__ void cvt_store(unsigned int* __restrict__ dst,
                                          const float4 r[4], float s) {
#pragma unroll
    for (int i = 0; i < 4; ++i) {
        dst[2 * i + 0] = pkh(r[i].x * s, r[i].y * s);
        dst[2 * i + 1] = pkh(r[i].z * s, r[i].w * s);
    }
}

__global__ __launch_bounds__(256)
void fp8lin_wmma_kernel(const float* __restrict__ A,     // [M_DIM, K_DIM]
                        const float* __restrict__ W,     // [N_DIM, K_DIM]
                        const float* __restrict__ S,     // [N_DIM/128, K_DIM/128]
                        float* __restrict__ C) {         // [M_DIM, N_DIM]
    __shared__ unsigned int Ash[2][BM * LDS_ROW_U];      // f16 tiles, padded rows
    __shared__ unsigned int Bsh[2][BN * LDS_ROW_U];

    const unsigned int tid  = threadIdx.x;
    const unsigned int lane = tid & 31u;
    const unsigned int wave = tid >> 5;          // 0..7
    const unsigned int wm   = wave & 3u;         // wave row group (M)
    const unsigned int wn   = wave >> 2;         // wave col group (N)
    const unsigned int lm   = lane & 15u;        // row/col within 16
    const unsigned int lhi  = lane >> 4;         // half-wave select

    const unsigned int n0 = blockIdx.x * BN;
    const unsigned int m0 = blockIdx.y * BM;

    // Loader mapping: 256 threads cover 128 rows x 32 K per tile (16 floats/thread).
    const unsigned int lr = tid >> 1;            // row 0..127
    const unsigned int lh = (tid & 1u) * 16u;    // K sub-offset 0 or 16

    const float* Aptr = A + (size_t)(m0 + lr) * K_DIM + lh;
    const float* Wptr = W + (size_t)(n0 + lr) * K_DIM + lh;
    const unsigned int ldst = lr * LDS_ROW_U + (lh >> 1);   // uint offset in LDS tile

    v8f acc[2][4];
#pragma unroll
    for (int mi = 0; mi < 2; ++mi)
#pragma unroll
        for (int ni = 0; ni < 4; ++ni)
            acc[mi][ni] = (v8f){0.f, 0.f, 0.f, 0.f, 0.f, 0.f, 0.f, 0.f};

    // ---- prologue: stage k-tile 0 ----
    {
        float4 ra[4], rb[4];
        load16(Aptr, ra);
        load16(Wptr, rb);
        const float s0 = S[blockIdx.x * SN + 0];
        cvt_store(&Ash[0][ldst], ra, 1.0f);
        cvt_store(&Bsh[0][ldst], rb, s0);
    }
    __syncthreads();

    for (unsigned int kt = 0; kt < KTILES; ++kt) {
        const unsigned int p = kt & 1u;
        const bool has_next = (kt + 1u < KTILES);

        // Issue next tile's global loads early (overlap with WMMA below).
        float4 ra[4], rb[4];
        float s_next = 0.f;
        if (has_next) {
            load16(Aptr + (size_t)(kt + 1u) * BK, ra);
            load16(Wptr + (size_t)(kt + 1u) * BK, rb);
            s_next = S[blockIdx.x * SN + ((kt + 1u) >> 2)];
        }
        if (kt + 2u < KTILES) {
            __builtin_prefetch(Aptr + (size_t)(kt + 2u) * BK, 0, 1);  // global_prefetch
            __builtin_prefetch(Wptr + (size_t)(kt + 2u) * BK, 0, 1);
        }

        // ---- gather WMMA fragments from LDS (documented CDNA5 layouts) ----
        // A 16x32 f16: lanes 0-15 row m K{0-7,16-23}; lanes 16-31 row m K{8-15,24-31}.
        FragU afrag[2];
#pragma unroll
        for (int mi = 0; mi < 2; ++mi) {
            const unsigned int row = wm * 32u + (unsigned)mi * 16u + lm;
            const unsigned int rb_ = row * LDS_ROW_U;
            afrag[mi].q[0] = *(const uint4*)&Ash[p][rb_ + 4u * lhi];
            afrag[mi].q[1] = *(const uint4*)&Ash[p][rb_ + 8u + 4u * lhi];
        }
        // B 32x16 f16: lane = column n; lanes 0-15 hold K{0-7,16-23}, lanes 16-31 hold K{8-15,24-31}.
        FragU bfrag[4];
#pragma unroll
        for (int ni = 0; ni < 4; ++ni) {
            const unsigned int row = wn * 64u + (unsigned)ni * 16u + lm;
            const unsigned int rb_ = row * LDS_ROW_U;
            bfrag[ni].q[0] = *(const uint4*)&Bsh[p][rb_ + 4u * lhi];
            bfrag[ni].q[1] = *(const uint4*)&Bsh[p][rb_ + 8u + 4u * lhi];
        }

        // ---- 8x v_wmma_f32_16x16x32_f16 ----
#pragma unroll
        for (int mi = 0; mi < 2; ++mi)
#pragma unroll
            for (int ni = 0; ni < 4; ++ni)
                acc[mi][ni] = __builtin_amdgcn_wmma_f32_16x16x32_f16(
                    /*neg_a=*/false, afrag[mi].v,
                    /*neg_b=*/false, bfrag[ni].v,
                    /*c_mod=*/(short)0, acc[mi][ni],
                    /*reuse_a=*/false, /*reuse_b=*/false);

        // ---- convert + stage next tile into the other buffer ----
        if (has_next) {
            cvt_store(&Ash[p ^ 1u][ldst], ra, 1.0f);
            cvt_store(&Bsh[p ^ 1u][ldst], rb, s_next);
        }
        __syncthreads();
    }

    // ---- epilogue: C layout — VGPR v: lanes 0-15 => (M=v, N=lane), lanes 16-31 => (M=v+8) ----
#pragma unroll
    for (int mi = 0; mi < 2; ++mi) {
#pragma unroll
        for (int ni = 0; ni < 4; ++ni) {
            const unsigned int nCol = n0 + wn * 64u + (unsigned)ni * 16u + lm;
#pragma unroll
            for (int v = 0; v < 8; ++v) {
                const unsigned int mRow = m0 + wm * 32u + (unsigned)mi * 16u + (unsigned)v + 8u * lhi;
                C[(size_t)mRow * N_DIM + nCol] = acc[mi][ni][v];
            }
        }
    }
}

extern "C" void kernel_launch(void* const* d_in, const int* in_sizes, int n_in,
                              void* d_out, int out_size, void* d_ws, size_t ws_size,
                              hipStream_t stream) {
    (void)in_sizes; (void)n_in; (void)out_size; (void)d_ws; (void)ws_size;
    const float* input  = (const float*)d_in[0];   // [1,512,7168]
    const float* weight = (const float*)d_in[1];   // [18432,7168]
    const float* scale  = (const float*)d_in[2];   // [144,56]
    float* out = (float*)d_out;                    // [1,512,18432]

    dim3 grid(N_DIM / BN, M_DIM / BM);             // (144, 4)
    dim3 block(256);
    fp8lin_wmma_kernel<<<grid, block, 0, stream>>>(input, weight, scale, out);
}
